// ConjugateModule_78022375899287
// MI455X (gfx1250) — matrix operand order repaired and checked by reading, not compiled, not run on hardware
//
#include <hip/hip_runtime.h>

typedef __attribute__((ext_vector_type(16))) _Float16 v16h;
typedef __attribute__((ext_vector_type(8)))  _Float16 v8h;
typedef __attribute__((ext_vector_type(8)))  float    v8f;
typedef unsigned int v4u __attribute__((ext_vector_type(4)));
typedef int          v8i __attribute__((ext_vector_type(8)));
typedef int          v4i __attribute__((ext_vector_type(4)));

#define Bb 4
#define Cc 64
#define Nn 131072
#define Ss 8
#define NT 256          // columns per workgroup in WMMA kernels (== blockDim)
#define CH 512          // columns per workgroup in updates kernel
#define XSTR 516        // k3 LDS row stride (floats): 16B-aligned rows, <=2-way conflicts
#define WSTR 72         // f16 row/column stride: 144B = 9*16B aligned, 2-way max conflicts

// workspace layout (in floats)
#define WS_SLOTSN   0          // 512   : l2-normalized slot_init (C,S)
#define WS_MX       512        // 32    : per-(b,s) row max
#define WS_SM       544        // 32    : per-(b,s) row expsum
#define WS_UPD      576        // 2048  : updates (b,S,C)
#define WS_ST       2624       // 2048  : slots_t  (b,S,C)
#define WS_SF       4672       // 2048  : slots_feat (b,S,C)
#define WS_SIM      8192       // B*S*N : sim rows

// ---- CDNA5 async global->LDS DMA (ISA 15.18.3, ASYNCcnt) ----
__device__ inline unsigned ldsOff(const void* p) {
  return (unsigned)(uintptr_t)(const __attribute__((address_space(3))) void*)p;
}
__device__ inline void asyncLoadB128(unsigned lds, const void* g) {
  asm volatile("global_load_async_to_lds_b128 %0, %1, off"
               :: "v"(lds), "v"((unsigned long long)(uintptr_t)g) : "memory");
}
__device__ inline void asyncWait0() {
  asm volatile("s_wait_asynccnt 0x0" ::: "memory");
}

// ---- CDNA5 Tensor Data Mover (ISA ch.8 D#; TENSORcnt) ----
#define HAVE_TDM __has_builtin(__builtin_amdgcn_tensor_load_to_lds)

// Dense 2D tile load: tileH rows x tileW elems (f32), global row stride
// rowStride elems, destination LDS contiguous row-major (tileW*4 bytes/row).
__device__ inline void tdmLoadTile2D(unsigned ldsByteAddr, const void* gAddr,
                                     unsigned tileW, unsigned tileH,
                                     unsigned tensorW, unsigned tensorH,
                                     unsigned rowStride) {
#if HAVE_TDM
  unsigned long long ga = (unsigned long long)(uintptr_t)gAddr;
  v4u g0;
  g0[0] = 1u;                                        // count=1, user descriptor
  g0[1] = ldsByteAddr;                               // lds_addr [63:32]
  g0[2] = (unsigned)(ga & 0xffffffffu);              // global_addr[31:0]
  g0[3] = (unsigned)((ga >> 32) & 0x01ffffffu)       // global_addr[56:32]
        | (2u << 30);                                // type=2 ("image")
  v8i g1;
  g1[0] = (int)(2u << 16);                           // wg_mask=0, data_size=4B
  g1[1] = (int)((tensorW & 0xffffu) << 16);          // tensor_dim0[15:0]
  g1[2] = (int)((tensorW >> 16) & 0xffffu)           // tensor_dim0[31:16]
        | (int)((tensorH & 0xffffu) << 16);          // tensor_dim1[15:0]
  g1[3] = (int)((tensorH >> 16) & 0xffffu)           // tensor_dim1[31:16]
        | (int)((tileW & 0xffffu) << 16);            // tile_dim0
  g1[4] = (int)(tileH & 0xffffu);                    // tile_dim1 (tile_dim2=0)
  g1[5] = (int)rowStride;                            // tensor_dim0_stride[31:0]
  g1[6] = 0;                                         // stride[47:32], dim1_stride
  g1[7] = 0;
  v4i z4 = {0, 0, 0, 0};
#if defined(__clang_major__) && __clang_major__ >= 23
  v8i z8 = {0, 0, 0, 0, 0, 0, 0, 0};
  __builtin_amdgcn_tensor_load_to_lds(g0, g1, z4, z4, z8, 0);
#else
  __builtin_amdgcn_tensor_load_to_lds(g0, g1, z4, z4, 0);
#endif
#else
  (void)ldsByteAddr; (void)gAddr; (void)tileW; (void)tileH;
  (void)tensorW; (void)tensorH; (void)rowStride;
#endif
}

// ---- WMMA fragment loaders (ISA 7.12.2 16-bit layouts, wave32) ----
__device__ inline v16h load_a72(const _Float16* W, int m0, int k0, int lane) {
  int m = m0 + (lane & 15);
  int hi = lane >> 4;
  v8h lo = *(const v8h*)&W[m * WSTR + k0 + 8 * hi];
  v8h hh = *(const v8h*)&W[m * WSTR + k0 + 16 + 8 * hi];
  v16h r;
#pragma unroll
  for (int e = 0; e < 8; ++e) { r[e] = lo[e]; r[8 + e] = hh[e]; }
  return r;
}
__device__ inline v16h load_b72(const _Float16* X, int k0, int nb, int lane) {
  int n = nb + (lane & 15);
  int hi = lane >> 4;
  v8h lo = *(const v8h*)&X[n * WSTR + k0 + 8 * hi];
  v8h hh = *(const v8h*)&X[n * WSTR + k0 + 16 + 8 * hi];
  v16h r;
#pragma unroll
  for (int e = 0; e < 8; ++e) { r[e] = lo[e]; r[8 + e] = hh[e]; }
  return r;
}

// ---- K0: l2-normalize slot_init over channels ----
__global__ void k0_slotsnorm(const float* __restrict__ slot_init, float* __restrict__ ws) {
  __shared__ float nrm[Ss];
  int t = threadIdx.x;
  if (t < Ss) {
    float acc = 0.f;
    for (int c = 0; c < Cc; ++c) { float v = slot_init[c * Ss + t]; acc += v * v; }
    nrm[t] = fmaxf(sqrtf(acc), 1e-12f);
  }
  __syncthreads();
  for (int i = t; i < Cc * Ss; i += blockDim.x)
    ws[WS_SLOTSN + i] = slot_init[i] / nrm[i % Ss];
}

// ---- K1: sim[b,s,n] = slots_norm^T @ (x / ||x||) via WMMA (S padded to 16) ----
__global__ void k1_sim(const float* __restrict__ x, float* __restrict__ ws) {
  extern __shared__ char smem[];
  float*    xf = (float*)smem;                  // Cc*NT f32 (TDM tile)
  _Float16* xh = (_Float16*)(xf + Cc * NT);     // NT*WSTR col-major, normalized f16
  _Float16* sA = xh + NT * WSTR;                // 16*WSTR padded slots_norm^T
  int b = blockIdx.y, t = threadIdx.x;
  int n0 = blockIdx.x * NT;
  const float* xb = x + (size_t)b * Cc * Nn + n0;

  unsigned xfOff = ldsOff(xf);
#if HAVE_TDM
  if ((t >> 5) == 0)                             // one TDM DMA for the whole tile
    tdmLoadTile2D(xfOff, xb, NT, Cc, Nn, Cc, Nn);
#else
#pragma unroll
  for (int j = 0; j < (Cc * NT / 4) / 256; ++j) {
    int i = t + j * 256;
    int c = i >> 6, q = (i & 63) * 4;
    asyncLoadB128(xfOff + (unsigned)(c * NT + q) * 4, xb + (size_t)c * Nn + q);
  }
#endif
  for (int i = t; i < 16 * Cc; i += 256) {       // A[m][k] = m<8 ? slotsN[k,s=m] : 0
    int m = i >> 6, k = i & 63;
    sA[m * WSTR + k] = (m < Ss) ? (_Float16)ws[WS_SLOTSN + k * Ss + m] : (_Float16)0.f;
  }
#if HAVE_TDM
  if ((t >> 5) == 0) __builtin_amdgcn_s_wait_tensorcnt(0);
#else
  asyncWait0();
#endif
  __syncthreads();
  {                                              // per-column inv-norm + f16 convert
    int n = t;
    float nrm2 = 0.f;
#pragma unroll
    for (int c = 0; c < Cc; ++c) { float v = xf[c * NT + n]; nrm2 += v * v; }
    float inv = 1.f / fmaxf(sqrtf(nrm2), 1e-12f);
#pragma unroll
    for (int c8 = 0; c8 < Cc; c8 += 8) {
      v8h h;
#pragma unroll
      for (int e = 0; e < 8; ++e) h[e] = (_Float16)(xf[(c8 + e) * NT + n] * inv);
      *(v8h*)&xh[n * WSTR + c8] = h;
    }
  }
  __syncthreads();
  int wid = t >> 5, lane = t & 31;
  int hi = lane >> 4, nl = lane & 15;
  v16h a0 = load_a72(sA, 0, 0, lane);
  v16h a1 = load_a72(sA, 0, 32, lane);
#pragma unroll
  for (int nbi = 0; nbi < 2; ++nbi) {            // 16 N-blocks over 8 waves
    int nb = (wid * 2 + nbi) * 16;
    v16h b0 = load_b72(xh, 0, nb, lane);
    v16h b1 = load_b72(xh, 32, nb, lane);
    v8f acc = {};
    acc = __builtin_amdgcn_wmma_f32_16x16x32_f16(false, a0, false, b0, (short)0, acc, false, false);
    acc = __builtin_amdgcn_wmma_f32_16x16x32_f16(false, a1, false, b1, (short)0, acc, false, false);
    if (hi == 0) {                               // rows M=0..7 are the real slots
#pragma unroll
      for (int r = 0; r < 8; ++r)
        ws[WS_SIM + (size_t)(b * Ss + r) * Nn + n0 + nb + nl] = acc[r];
    }
  }
}

// ---- K2: per-(b,s) softmax row stats over N; also zero updates ----
__global__ void k2_rowstats(float* __restrict__ ws) {
  __shared__ float red[256];
  int row = blockIdx.x, t = threadIdx.x;
  const float* sim = ws + WS_SIM + (size_t)row * Nn;
  float m = -3.4e38f;
  for (int i = t; i < Nn; i += 256) m = fmaxf(m, sim[i]);
  red[t] = m; __syncthreads();
  for (int o = 128; o > 0; o >>= 1) { if (t < o) red[t] = fmaxf(red[t], red[t + o]); __syncthreads(); }
  float mx = red[0]; __syncthreads();
  float s = 0.f;
  for (int i = t; i < Nn; i += 256) s += __expf(sim[i] - mx);
  red[t] = s; __syncthreads();
  for (int o = 128; o > 0; o >>= 1) { if (t < o) red[t] += red[t + o]; __syncthreads(); }
  if (t == 0) { ws[WS_MX + row] = mx; ws[WS_SM + row] = red[0]; }
  if (t < Cc) ws[WS_UPD + row * Cc + t] = 0.f;
}

// ---- K3: updates[b,s,c] += sum_n x[b,c,n] * attn[b,s,n] (per-lane async DMA) ----
__global__ void k3_updates(const float* __restrict__ x, float* __restrict__ ws) {
  extern __shared__ float sm3[];
  float* attnL = sm3;               // Ss*CH
  float* xL = sm3 + Ss * CH;        // Cc*XSTR (padded, 16B-aligned rows)
  int b = blockIdx.y, t = threadIdx.x;
  int n0 = blockIdx.x * CH;
  unsigned xOff = ldsOff(xL);
#pragma unroll
  for (int j = 0; j < (Cc * CH / 4) / 256; ++j) {  // 32 async b128 per thread
    int i = t + j * 256;
    int c = i >> 7, q = (i & 127) * 4;
    asyncLoadB128(xOff + (unsigned)(c * XSTR + q) * 4,
                  &x[((size_t)b * Cc + c) * Nn + n0 + q]);
  }
  for (int i = t; i < Ss * CH; i += 256) {
    int s = i / CH, nn = i % CH;
    int row = b * Ss + s;
    float v = ws[WS_SIM + (size_t)row * Nn + n0 + nn];
    attnL[i] = __expf(v - ws[WS_MX + row]) / ws[WS_SM + row];
  }
  asyncWait0();
  __syncthreads();
  for (int p = t; p < Ss * Cc; p += 256) {
    int s = p >> 6, c = p & 63;
    float acc = 0.f;
    const float* xr = xL + c * XSTR;
    const float* ar = attnL + s * CH;
#pragma unroll 8
    for (int nn = 0; nn < CH; ++nn) acc += xr[nn] * ar[nn];
    atomicAdd(&ws[WS_UPD + (b * Ss + s) * Cc + c], acc);
  }
}

// ---- K4: slot_proj + LayerNorm + tiny MHSA (S=8, 4 heads) + out_proj + s2f ----
__global__ void k4_slots(const float* __restrict__ spw, const float* __restrict__ spb,
                         const float* __restrict__ lng, const float* __restrict__ lnb,
                         const float* __restrict__ ipw, const float* __restrict__ ipb,
                         const float* __restrict__ opw, const float* __restrict__ opb,
                         const float* __restrict__ s2fw, const float* __restrict__ s2fb,
                         float* __restrict__ ws) {
  __shared__ float upd[512], st[512], sn4[512], qkv[1536], tmp4[512], sc[256], mu[8], rstd[8];
  int b = blockIdx.x, t = threadIdx.x;
  for (int i = t; i < 512; i += 256) upd[i] = ws[WS_UPD + b * 512 + i];
  __syncthreads();
  for (int i = t; i < 512; i += 256) {             // slots_t = updates @ W^T + b
    int s = i >> 6, o = i & 63;
    float acc = spb[o];
    for (int c = 0; c < Cc; ++c) acc += upd[s * 64 + c] * spw[o * 64 + c];
    st[i] = acc;
  }
  __syncthreads();
  if (t < 8) {                                     // LayerNorm stats per slot
    float m = 0.f; for (int c = 0; c < 64; ++c) m += st[t * 64 + c];
    m *= (1.f / 64.f);
    float v = 0.f; for (int c = 0; c < 64; ++c) { float d = st[t * 64 + c] - m; v += d * d; }
    mu[t] = m; rstd[t] = rsqrtf(v * (1.f / 64.f) + 1e-5f);
  }
  __syncthreads();
  for (int i = t; i < 512; i += 256) {
    int s = i >> 6, c = i & 63;
    sn4[i] = (st[i] - mu[s]) * rstd[s] * lng[c] + lnb[c];
  }
  __syncthreads();
  for (int i = t; i < 1536; i += 256) {            // qkv projection
    int s = i / 192, o = i % 192;
    float acc = ipb[o];
    for (int c = 0; c < 64; ++c) acc += sn4[s * 64 + c] * ipw[o * 64 + c];
    qkv[i] = acc;
  }
  __syncthreads();
  {                                                // scores (4 heads, 8x8), scale 1/4
    int h = t >> 6, q = (t >> 3) & 7, k = t & 7;
    float acc = 0.f;
    for (int d = 0; d < 16; ++d)
      acc += qkv[q * 192 + h * 16 + d] * qkv[k * 192 + 64 + h * 16 + d];
    sc[t] = acc * 0.25f;
  }
  __syncthreads();
  if (t < 32) {                                    // softmax over k
    int h = t >> 3, q = t & 7;
    float* r = sc + h * 64 + q * 8;
    float m = r[0]; for (int k = 1; k < 8; ++k) m = fmaxf(m, r[k]);
    float ssum = 0.f; for (int k = 0; k < 8; ++k) { r[k] = __expf(r[k] - m); ssum += r[k]; }
    float is = 1.f / ssum; for (int k = 0; k < 8; ++k) r[k] *= is;
  }
  __syncthreads();
  for (int i = t; i < 512; i += 256) {             // o = a @ v
    int s = i >> 6, ch = i & 63, h = ch >> 4, d = ch & 15;
    float acc = 0.f;
    for (int k = 0; k < 8; ++k) acc += sc[h * 64 + s * 8 + k] * qkv[k * 192 + 128 + h * 16 + d];
    tmp4[i] = acc;
  }
  __syncthreads();
  for (int i = t; i < 512; i += 256) {             // out_proj + residual -> slots_t
    int s = i >> 6, o = i & 63;
    float acc = opb[o];
    for (int c = 0; c < 64; ++c) acc += tmp4[s * 64 + c] * opw[o * 64 + c];
    float v = st[i] + acc;
    sn4[i] = v;
    ws[WS_ST + b * 512 + i] = v;
  }
  __syncthreads();
  for (int i = t; i < 512; i += 256) {             // slots_feat = slots_t @ s2f^T + b
    int s = i >> 6, o = i & 63;
    float acc = s2fb[o];
    for (int c = 0; c < 64; ++c) acc += sn4[s * 64 + c] * s2fw[o * 64 + c];
    ws[WS_SF + b * 512 + i] = acc;
  }
}

// ---- K5: fused feat-conv (WMMA) + slot softmax + broadcast + residual ----
__global__ void k5_fused(const float* __restrict__ x,
                         const float* __restrict__ feat_w, const float* __restrict__ feat_b,
                         const float* __restrict__ ws, float* __restrict__ out) {
  extern __shared__ char smem[];
  float*    xf    = (float*)smem;                    // Cc*NT f32 (TDM tile + residual)
  _Float16* xh    = (_Float16*)(xf + Cc * NT);       // NT*WSTR col-major f16
  _Float16* featH = xh + NT * WSTR;                  // NT*WSTR col-major f16
  _Float16* wh    = featH + NT * WSTR;               // Cc*WSTR row-major f16
  float*    sfeatL= (float*)(wh + Cc * WSTR);        // 512
  float*    stL   = sfeatL + 512;                    // 512
  float*    fbL   = stL + 512;                       // 64
  int b = blockIdx.y, t = threadIdx.x;
  int n0 = blockIdx.x * NT;
  const float* xb = x + (size_t)b * Cc * Nn + n0;

  unsigned xfOff = ldsOff(xf);
#if HAVE_TDM
  if ((t >> 5) == 0)                                 // one TDM DMA for the whole tile
    tdmLoadTile2D(xfOff, xb, NT, Cc, Nn, Cc, Nn);
#else
#pragma unroll
  for (int j = 0; j < (Cc * NT / 4) / 256; ++j) {
    int i = t + j * 256;
    int c = i >> 6, q = (i & 63) * 4;
    asyncLoadB128(xfOff + (unsigned)(c * NT + q) * 4, xb + (size_t)c * Nn + q);
  }
#endif
  for (int i = t; i < Cc * Cc; i += 256)             // overlap param staging with DMA
    wh[(i >> 6) * WSTR + (i & 63)] = (_Float16)feat_w[i];
  for (int i = t; i < 512; i += 256) {
    sfeatL[i] = ws[WS_SF + b * 512 + i];
    stL[i]    = ws[WS_ST + b * 512 + i];
  }
  if (t < Cc) fbL[t] = feat_b[t];
  if (n0 + NT < Nn)                                  // prefetch next x tile
    for (int c = 0; c < Cc; c += 8)
      __builtin_prefetch(xb + (size_t)c * Nn + NT + t, 0, 0);
#if HAVE_TDM
  if ((t >> 5) == 0) __builtin_amdgcn_s_wait_tensorcnt(0);
#else
  asyncWait0();
#endif
  __syncthreads();
  {                                                  // f32 -> f16, column-major
    int n = t;
#pragma unroll
    for (int c8 = 0; c8 < Cc; c8 += 8) {
      v8h h;
#pragma unroll
      for (int e = 0; e < 8; ++e) h[e] = (_Float16)xf[(c8 + e) * NT + n];
      *(v8h*)&xh[n * WSTR + c8] = h;
    }
  }
  __syncthreads();

  // feat = f16(feat_w) @ f16(x), f32 accum; 2-way interleaved accumulators
  int wid = t >> 5, lane = t & 31;
  int m0 = (wid & 3) * 16;
  int nbBase = (wid >> 2) * 8;
  int hi = lane >> 4, nl = lane & 15;
  v16h a0 = load_a72(wh, m0, 0, lane);
  v16h a1 = load_a72(wh, m0, 32, lane);
  float bias[8];
#pragma unroll
  for (int r = 0; r < 8; ++r) bias[r] = fbL[m0 + r + 8 * hi];
#pragma unroll
  for (int nbi = 0; nbi < 8; nbi += 2) {
    int nbA = (nbBase + nbi) * 16, nbB = nbA + 16;
    v16h b0 = load_b72(xh, 0, nbA, lane);
    v16h b1 = load_b72(xh, 32, nbA, lane);
    v16h b2 = load_b72(xh, 0, nbB, lane);
    v16h b3 = load_b72(xh, 32, nbB, lane);
    v8f accA = {}, accB = {};
    accA = __builtin_amdgcn_wmma_f32_16x16x32_f16(false, a0, false, b0, (short)0, accA, false, false);
    accB = __builtin_amdgcn_wmma_f32_16x16x32_f16(false, a0, false, b2, (short)0, accB, false, false);
    accA = __builtin_amdgcn_wmma_f32_16x16x32_f16(false, a1, false, b1, (short)0, accA, false, false);
    accB = __builtin_amdgcn_wmma_f32_16x16x32_f16(false, a1, false, b3, (short)0, accB, false, false);
    v8h hA, hB;
#pragma unroll
    for (int r = 0; r < 8; ++r) {
      hA[r] = (_Float16)(accA[r] + bias[r]);
      hB[r] = (_Float16)(accB[r] + bias[r]);
    }
    *(v8h*)&featH[(nbA + nl) * WSTR + m0 + 8 * hi] = hA;   // one ds_store_b128
    *(v8h*)&featH[(nbB + nl) * WSTR + m0 + 8 * hi] = hB;
  }
  __syncthreads();

  // per-column: sim2 over slots, softmax over S=8 (column-local), broadcast + residual
  {
    int n = t;
    float a2[Ss];
#pragma unroll
    for (int s = 0; s < Ss; ++s) a2[s] = 0.f;
#pragma unroll 8
    for (int c = 0; c < Cc; ++c) {
      float f = (float)featH[n * WSTR + c];
#pragma unroll
      for (int s = 0; s < Ss; ++s) a2[s] += sfeatL[s * 64 + c] * f;
    }
    float m = a2[0];
#pragma unroll
    for (int s = 1; s < Ss; ++s) m = fmaxf(m, a2[s]);
    float ssum = 0.f;
#pragma unroll
    for (int s = 0; s < Ss; ++s) { a2[s] = __expf(a2[s] - m); ssum += a2[s]; }
    float is = 1.f / ssum;
#pragma unroll
    for (int s = 0; s < Ss; ++s) a2[s] *= is;
    for (int c = 0; c < Cc; ++c) {
      float o = 0.f;
#pragma unroll
      for (int s = 0; s < Ss; ++s) o += stL[s * 64 + c] * a2[s];
      out[((size_t)b * Cc + c) * Nn + n0 + n] = o + xf[c * NT + n];
    }
  }
}

extern "C" void kernel_launch(void* const* d_in, const int* in_sizes, int n_in,
                              void* d_out, int out_size, void* d_ws, size_t ws_size,
                              hipStream_t stream) {
  (void)in_sizes; (void)n_in; (void)out_size; (void)ws_size;
  const float* x    = (const float*)d_in[0];
  const float* sini = (const float*)d_in[1];
  const float* spw  = (const float*)d_in[2];
  const float* spb  = (const float*)d_in[3];
  const float* lng  = (const float*)d_in[4];
  const float* lnb  = (const float*)d_in[5];
  const float* ipw  = (const float*)d_in[6];
  const float* ipb  = (const float*)d_in[7];
  const float* opw  = (const float*)d_in[8];
  const float* opb  = (const float*)d_in[9];
  const float* fw   = (const float*)d_in[10];
  const float* fb   = (const float*)d_in[11];
  const float* s2fw = (const float*)d_in[12];
  const float* s2fb = (const float*)d_in[13];
  float* out = (float*)d_out;
  float* ws  = (float*)d_ws;

  k0_slotsnorm<<<1, 64, 0, stream>>>(sini, ws);
  size_t lds1 = (size_t)(Cc * NT) * 4 + (size_t)(NT * WSTR) * 2 + (size_t)(16 * WSTR) * 2;
  k1_sim<<<dim3(Nn / NT, Bb), 256, lds1, stream>>>(x, ws);
  k2_rowstats<<<Bb * Ss, 256, 0, stream>>>(ws);
  size_t lds3 = (size_t)(Ss * CH + Cc * XSTR) * sizeof(float);
  k3_updates<<<dim3(Nn / CH, Bb), 256, lds3, stream>>>(x, ws);
  k4_slots<<<Bb, 256, 0, stream>>>(spw, spb, lng, lnb, ipw, ipb, opw, opb, s2fw, s2fb, ws);
  size_t lds5 = (size_t)(Cc * NT) * 4 + (size_t)(NT * WSTR) * 2 * 2 +
                (size_t)(Cc * WSTR) * 2 + (size_t)(512 + 512 + 64) * 4;
  k5_fused<<<dim3(Nn / NT, Bb), 256, lds5, stream>>>(x, fw, fb, ws, out);
}
